// RAFTStereoHuman_54417235640940
// MI455X (gfx1250) — matrix operand order repaired and checked by reading, not compiled, not run on hardware
//
#include <hip/hip_runtime.h>
#include <math.h>

// ---------------- WMMA types ----------------
typedef __attribute__((ext_vector_type(16))) _Float16 v16h;
typedef __attribute__((ext_vector_type(8)))  float    v8f;
typedef __attribute__((ext_vector_type(4)))  int      v4i;

// CDNA5 16-bit WMMA fragment maps (wave32):
// A (16x32, M=lane&15): element K lives at lane = M + 16*((K>>3)&1), half j = ((K>>4)<<3)|(K&7)
// B (32x16, N=lane&15): element K lives at lane = N + 16*((K>>3)&1), half j = ((K>>4)<<3)|(K&7)
// C/D (16x16 f32): VGPR r -> M = r + ((lane>>4)<<3), N = lane&15

__device__ __forceinline__ void fdivmod(int x, int d, float inv, int& q, int& r) {
    q = (int)((float)x * inv);
    r = x - q * d;
    if (r < 0) { --q; r += d; }
    else if (r >= d) { ++q; r -= d; }
}

// ---- weight pre-conversion: f32 [Cout,Ktot] -> f16 [Cout,Kpad] (zero padded) ----
__global__ __launch_bounds__(256)
void cvt_w_f16(const float* __restrict__ w, _Float16* __restrict__ o,
               int Cout, int Ktot, int Kpad) {
    int i = blockIdx.x * blockDim.x + threadIdx.x;
    if (i >= Cout * Kpad) return;
    int co = i / Kpad, k = i - co * Kpad;
    o[i] = (k < Ktot) ? (_Float16)w[(size_t)co * Ktot + k] : (_Float16)0.f;
}

// =====================================================================
// all-pairs correlation: corr[b,h,w1,w2] = sum_d f1*f2 / sqrt(D)
// =====================================================================
template <int WP>
__global__ __launch_bounds__(256)
void corr_wmma(const float* __restrict__ f1, const float* __restrict__ f2,
               float* __restrict__ corr, int D, int H, int W)
{
    constexpr int NSUB = WP / 16;
    constexpr int CH = WP / 8;
    __shared__ v16h aF[32];
    __shared__ v16h bF[NSUB][32];

    const int tid = threadIdx.x;
    const int lane = tid & 31, wid = tid >> 5;
    const int half = lane >> 4, l15 = lane & 15;
    const int bh = blockIdx.y;
    const int b = bh / H, h = bh - b * H;
    const int m_base = blockIdx.x * 16;
    const int HW = H * W;

    v8f acc = {};
    const float* f1p = f1 + ((size_t)b * D * H + h) * W;
    const float* f2p = f2 + ((size_t)b * D * H + h) * W;
    const int bK = tid >> 3;
    const int n0 = (tid & 7) * CH;

    for (int d0 = 0; d0 < D; d0 += 32) {
        float va[2], vb[CH];
        #pragma unroll
        for (int u = 0; u < 2; ++u) {
            int e = tid * 2 + u;
            int m = e & 15, K = e >> 4;
            int w1 = m_base + m;
            int w1c = w1 < W ? w1 : W - 1;
            va[u] = f1p[(size_t)(d0 + K) * HW + w1c];
        }
        #pragma unroll
        for (int u = 0; u < CH; ++u) {
            int n = n0 + u;
            int nc = n < W ? n : W - 1;
            vb[u] = f2p[(size_t)(d0 + bK) * HW + nc];
        }
        #pragma unroll
        for (int u = 0; u < 2; ++u) {
            int e = tid * 2 + u;
            int m = e & 15, K = e >> 4;
            float v = (m_base + m) < W ? va[u] : 0.f;
            int fl = m + (((K >> 3) & 1) << 4);
            int fj = ((K >> 4) << 3) | (K & 7);
            ((_Float16*)&aF[fl])[fj] = (_Float16)v;
        }
        #pragma unroll
        for (int u = 0; u < CH; ++u) {
            int n = n0 + u;
            float v = n < W ? vb[u] : 0.f;
            int fl = (n & 15) + (((bK >> 3) & 1) << 4);
            ((_Float16*)&bF[n >> 4][fl])[((bK >> 4) << 3) | (bK & 7)] = (_Float16)v;
        }
        __syncthreads();
        if (wid < NSUB) {
            v16h af = aF[lane];
            v16h bf = bF[wid][lane];
            acc = __builtin_amdgcn_wmma_f32_16x16x32_f16(false, af, false, bf,
                                                         (short)0, acc, false, false);
        }
        __syncthreads();
    }
    if (wid < NSUB) {
        const float s = rsqrtf((float)D);
        #pragma unroll
        for (int r = 0; r < 8; ++r) {
            int m = m_base + r + (half << 3);
            int n = wid * 16 + l15;
            if (m < W && n < W)
                corr[((size_t)(b * H + h) * W + m) * W + n] = acc[r] * s;
        }
    }
}

// =====================================================================
// implicit-GEMM conv with pre-converted f16 weights (rows padded to 32).
// block 256 computes 64(Cout) x 64(spatial); LDS double-buffered.
// act: 0=none 1=relu 2=sigmoid 3=tanh
// =====================================================================
__global__ __launch_bounds__(256)
void conv_wmma(const float* __restrict__ in0, int C0,
               const float* __restrict__ in1, int C1,
               const float* __restrict__ in2, int C2,
               const _Float16* __restrict__ wgt, const float* __restrict__ bias,
               const float* __restrict__ addv, float* __restrict__ out,
               int Bn, int H, int W, int Cout, int ostride,
               int ks, int pad, int act, float oscale)
{
    __shared__ v16h aF[2][4][32];
    __shared__ v16h bF[2][4][32];

    const int tid = threadIdx.x;
    const int lane = tid & 31, wid = tid >> 5;
    const int wm = wid & 3, wn = wid >> 2;
    const int half = lane >> 4, l15 = lane & 15;

    const int Cin = C0 + C1 + C2;
    const int KK = ks * ks;
    const int Ktot = Cin * KK;
    const int Kpad = (Ktot + 31) & ~31;
    const int NP = Bn * H * W, HW = H * W;
    const int m_base = blockIdx.x * 64, n_base = blockIdx.y * 64;
    const float invKK = 1.f / (float)KK, invks = 1.f / (float)ks;

    // ---- A invariants: 8 consecutive k of one co row -> 16B LDS chunk ----
    const int am = tid >> 2;                    // 0..63
    const int aK0 = (tid * 8) & 31;             // {0,8,16,24}
    int aco = m_base + am; if (aco >= Cout) aco = Cout - 1;
    const size_t a_rowbase = (size_t)aco * Kpad;
    const int a_fl = (am & 15) + (((aK0 >> 3) & 1) << 4);
    const int a_off = ((am >> 4) * 32 + a_fl) * 16 + ((aK0 >> 4) << 3);

    // ---- B invariants: one K row, 8 consecutive n ----
    const int bK = tid >> 3;                    // 0..31, constant per thread
    const int nn0 = (tid * 8) & 63;
    const int n0 = n_base + nn0;
    int b_bb[8], b_y[8], b_x[8];
    bool b_nok[8];
    #pragma unroll
    for (int u = 0; u < 8; ++u) {
        int n = n0 + u;
        b_nok[u] = n < NP;
        int ncl = b_nok[u] ? n : 0;
        int bb = ncl / HW;
        int rem = ncl - bb * HW;
        b_bb[u] = bb; b_y[u] = rem / W; b_x[u] = rem - b_y[u] * W;
    }
    const bool rowfast = (n0 + 7 < NP) && (b_x[0] + 7 < W);
    int b_off[8];
    #pragma unroll
    for (int u = 0; u < 8; ++u) {
        int nn = nn0 + u;
        int fl = (nn & 15) + (((bK >> 3) & 1) << 4);
        b_off[u] = ((nn >> 4) * 32 + fl) * 16 + (((bK >> 4) << 3) | (bK & 7));
    }

    auto stage = [&](int k0, int p) {
        // A: single 16B load (pad zeros handle k overrun; bad co rows discarded later)
        v4i aw = *(const v4i*)(wgt + a_rowbase + (k0 + aK0));
        // B: one divmod per thread, then 8 gathers (fast: contiguous row span)
        int kk = k0 + bK;
        bool kok = kk < Ktot;
        int kc = kok ? kk : 0;
        int ci, rr, ky, kx;
        fdivmod(kc, KK, invKK, ci, rr);
        fdivmod(rr, ks, invks, ky, kx);
        const float* src; int cs, cn;
        if (ci < C0)           { src = in0; cs = ci;           cn = C0; }
        else if (ci < C0 + C1) { src = in1; cs = ci - C0;      cn = C1; }
        else                   { src = in2; cs = ci - C0 - C1; cn = C2; }
        float vb[8];
        bool inb[8];
        int iy = b_y[0] + ky - pad;
        int ix0 = b_x[0] + kx - pad;
        if (rowfast && kok && iy >= 0 && iy < H && ix0 >= 0 && ix0 + 7 < W) {
            const float* bp = src + ((size_t)(b_bb[0] * cn + cs) * H + iy) * W + ix0;
            #pragma unroll
            for (int u = 0; u < 8; ++u) { vb[u] = bp[u]; inb[u] = true; }
        } else {
            #pragma unroll
            for (int u = 0; u < 8; ++u) {
                int iyu = b_y[u] + ky - pad, ixu = b_x[u] + kx - pad;
                inb[u] = kok && b_nok[u] && iyu >= 0 && iyu < H && ixu >= 0 && ixu < W;
                int iyc = iyu < 0 ? 0 : (iyu >= H ? H - 1 : iyu);
                int ixc = ixu < 0 ? 0 : (ixu >= W ? W - 1 : ixu);
                vb[u] = src[((size_t)(b_bb[u] * cn + cs) * H + iyc) * W + ixc];
            }
        }
        // stores
        *(v4i*)(((_Float16*)&aF[p][0][0]) + a_off) = aw;
        _Float16* bpl = (_Float16*)&bF[p][0][0];
        #pragma unroll
        for (int u = 0; u < 8; ++u)
            bpl[b_off[u]] = (_Float16)(inb[u] ? vb[u] : 0.f);
    };

    v8f acc0 = {}, acc1 = {};
    stage(0, 0);
    int cur = 0;
    for (int k0 = 0; k0 < Ktot; k0 += 32) {
        __syncthreads();
        v16h af = aF[cur][wm][lane];
        v16h b0 = bF[cur][wn * 2 + 0][lane];
        v16h b1 = bF[cur][wn * 2 + 1][lane];
        if (k0 + 32 < Ktot) stage(k0 + 32, cur ^ 1);
        acc0 = __builtin_amdgcn_wmma_f32_16x16x32_f16(false, af, false, b0,
                                                      (short)0, acc0, false, false);
        acc1 = __builtin_amdgcn_wmma_f32_16x16x32_f16(false, af, false, b1,
                                                      (short)0, acc1, false, false);
        cur ^= 1;
    }

    // ---- epilogue ----
    #pragma unroll
    for (int t = 0; t < 2; ++t) {
        v8f acc = (t == 0) ? acc0 : acc1;
        #pragma unroll
        for (int r = 0; r < 8; ++r) {
            int co = m_base + wm * 16 + r + (half << 3);
            int n = n_base + (wn * 2 + t) * 16 + l15;
            if (co < Cout && n < NP) {
                int bb = n / HW;
                int rem = n - bb * HW;
                int y = rem / W, x = rem - y * W;
                size_t oidx = ((size_t)(bb * ostride + co) * H + y) * W + x;
                float v = acc[r] + bias[co];
                if (addv) v += addv[oidx];
                if (act == 1)      v = fmaxf(v, 0.f);
                else if (act == 2) v = 1.f / (1.f + expf(-v));
                else if (act == 3) v = tanhf(v);
                out[oidx] = v * oscale;
            }
        }
    }
}

// ---------------- elementwise / support kernels ----------------
__global__ __launch_bounds__(256) void ew_tanh(const float* in, float* out, int n) {
    int i = blockIdx.x * blockDim.x + threadIdx.x;
    if (i < n) out[i] = tanhf(in[i]);
}
__global__ __launch_bounds__(256) void ew_mul(const float* a, const float* b, float* o, int n) {
    int i = blockIdx.x * blockDim.x + threadIdx.x;
    if (i < n) o[i] = a[i] * b[i];
}
__global__ __launch_bounds__(256) void gru_comb(float* h, const float* z, const float* q, int n) {
    int i = blockIdx.x * blockDim.x + threadIdx.x;
    if (i < n) h[i] = (1.f - z[i]) * h[i] + z[i] * q[i];
}
__global__ __launch_bounds__(256) void avgpair(const float* in, float* out, int outer, int Wl) {
    int i = blockIdx.x * blockDim.x + threadIdx.x;
    if (i >= outer * Wl) return;
    int o = i / Wl, w = i - o * Wl;
    out[i] = 0.5f * (in[(size_t)o * 2 * Wl + 2 * w] + in[(size_t)o * 2 * Wl + 2 * w + 1]);
}
__global__ __launch_bounds__(256) void init_cx(float* cx, int n, int W) {
    int i = blockIdx.x * blockDim.x + threadIdx.x;
    if (i < n) cx[i] = (float)(i % W);
}
__global__ __launch_bounds__(256)
void corr_lookup_k(const float* c0, const float* c1, const float* c2, const float* c3,
                   const float* cx, float* out, int Bn, int H, int W)
{
    int i = blockIdx.x * blockDim.x + threadIdx.x;
    int total = Bn * H * W;
    if (i >= total) return;
    int b = i / (H * W);
    int rem = i - b * H * W;
    int y = rem / W, x = rem - y * W;
    float xc = cx[i];
    const float* cs[4] = {c0, c1, c2, c3};
    int Wl = W;
    float inv = 1.f;
    for (int lv = 0; lv < 4; ++lv) {
        const float* row = cs[lv] + ((size_t)(b * H + y) * W + x) * Wl;
        float xs0 = xc * inv;
        for (int j = 0; j < 9; ++j) {
            float xs = xs0 + (float)(j - 4);
            float x0f = floorf(xs);
            float wf = xs - x0f;
            int i0 = (int)x0f, i1 = i0 + 1;
            float v0 = (i0 >= 0 && i0 < Wl) ? row[i0] : 0.f;
            float v1 = (i1 >= 0 && i1 < Wl) ? row[i1] : 0.f;
            out[((size_t)(b * 36 + lv * 9 + j) * H + y) * W + x] = v0 * (1.f - wf) + v1 * wf;
        }
        Wl >>= 1;
        inv *= 0.5f;
    }
}
__global__ __launch_bounds__(256)
void flow_k(const float* cx, float* flow, int Bn, int H, int W) {
    int i = blockIdx.x * blockDim.x + threadIdx.x;
    int total = Bn * H * W;
    if (i >= total) return;
    int b = i / (H * W);
    int rem = i - b * H * W;
    int x = rem % W;
    flow[(size_t)b * 2 * H * W + rem] = cx[i] - (float)x;
    flow[(size_t)b * 2 * H * W + H * W + rem] = 0.f;
}
__global__ __launch_bounds__(256)
void pool2x_k(const float* in, float* out, int Bn, int C, int Hi, int Wi) {
    int Ho = Hi / 2, Wo = Wi / 2;
    int total = Bn * C * Ho * Wo;
    int i = blockIdx.x * blockDim.x + threadIdx.x;
    if (i >= total) return;
    int bc = i / (Ho * Wo);
    int rem = i - bc * Ho * Wo;
    int yo = rem / Wo, xo = rem - yo * Wo;
    float s = 0.f;
    for (int dy = 0; dy < 3; ++dy)
        for (int dx = 0; dx < 3; ++dx) {
            int y = 2 * yo - 1 + dy, x = 2 * xo - 1 + dx;
            if (y >= 0 && y < Hi && x >= 0 && x < Wi)
                s += in[(size_t)bc * Hi * Wi + y * Wi + x];
        }
    out[i] = s * (1.f / 9.f);
}
__global__ __launch_bounds__(256)
void interp_k(const float* in, float* out, int Bn, int C, int Hi, int Wi, int Ho, int Wo) {
    int total = Bn * C * Ho * Wo;
    int i = blockIdx.x * blockDim.x + threadIdx.x;
    if (i >= total) return;
    int bc = i / (Ho * Wo);
    int rem = i - bc * Ho * Wo;
    int yo = rem / Wo, xo = rem - yo * Wo;
    float sy = (float)yo * ((float)(Hi - 1) / (float)(Ho - 1));
    float sx = (float)xo * ((float)(Wi - 1) / (float)(Wo - 1));
    int y0 = (int)floorf(sy); if (y0 > Hi - 2) y0 = Hi - 2; if (y0 < 0) y0 = 0;
    int x0 = (int)floorf(sx); if (x0 > Wi - 2) x0 = Wi - 2; if (x0 < 0) x0 = 0;
    float wy = sy - (float)y0, wx = sx - (float)x0;
    const float* p = in + (size_t)bc * Hi * Wi;
    float t0 = p[y0 * Wi + x0] * (1.f - wy) + p[(y0 + 1) * Wi + x0] * wy;
    float t1 = p[y0 * Wi + x0 + 1] * (1.f - wy) + p[(y0 + 1) * Wi + x0 + 1] * wy;
    out[i] = t0 * (1.f - wx) + t1 * wx;
}
__global__ __launch_bounds__(256)
void mf_tail_k(const float* cx, float* mf, int Bn, int H, int W) {
    int i = blockIdx.x * blockDim.x + threadIdx.x;
    int total = Bn * H * W;
    if (i >= total) return;
    int b = i / (H * W);
    int rem = i - b * H * W;
    int x = rem % W;
    mf[((size_t)(b * 128 + 126) * H) * W + rem] = cx[i] - (float)x;
    mf[((size_t)(b * 128 + 127) * H) * W + rem] = 0.f;
}
__global__ __launch_bounds__(256)
void coords_upd(float* cx, const float* df, int Bn, int H, int W) {
    int i = blockIdx.x * blockDim.x + threadIdx.x;
    int total = Bn * H * W;
    if (i >= total) return;
    int b = i / (H * W);
    int rem = i - b * H * W;
    cx[i] += df[(size_t)b * 2 * H * W + rem];
}
__global__ __launch_bounds__(256)
void upsample_k(const float* cx, const float* mask, float* out, int Bn, int H, int W) {
    int HF = H * 8, WF = W * 8;
    int total = Bn * HF * WF;
    int i = blockIdx.x * blockDim.x + threadIdx.x;
    if (i >= total) return;
    int b = i / (HF * WF);
    int rem = i - b * HF * WF;
    int Y = rem / WF, X = rem - Y * WF;
    int y = Y >> 3, fy = Y & 7, x = X >> 3, fx = X & 7;
    float wv[9], mx = -1e30f;
    for (int k = 0; k < 9; ++k) {
        wv[k] = mask[((size_t)(b * 576 + k * 64 + fy * 8 + fx) * H + y) * W + x];
        mx = fmaxf(mx, wv[k]);
    }
    float se = 0.f;
    for (int k = 0; k < 9; ++k) { wv[k] = expf(wv[k] - mx); se += wv[k]; }
    float accv = 0.f;
    for (int k = 0; k < 9; ++k) {
        int ny = y + k / 3 - 1, nx = x + k % 3 - 1;
        float pv = 0.f;
        if (ny >= 0 && ny < H && nx >= 0 && nx < W)
            pv = 8.f * (cx[(size_t)b * H * W + ny * W + nx] - (float)nx);
        accv += wv[k] * pv;
    }
    out[i] = accv / se;
}

// ---------------- host orchestration ----------------
extern "C" void kernel_launch(void* const* d_in, const int* in_sizes, int n_in,
                              void* d_out, int out_size, void* d_ws, size_t ws_size,
                              hipStream_t stream)
{
    (void)in_sizes; (void)n_in; (void)out_size; (void)ws_size;
    const int B = 2, D = 256, H = 48, W = 72, hd = 128;
    const int H2 = 24, W2 = 36, H4 = 12, W4 = 18;
    const int ITERS = 12;

    const float* fmap1 = (const float*)d_in[0];
    const float* fmap2 = (const float*)d_in[1];
    const float* net0_in = (const float*)d_in[2];
    const float* net1_in = (const float*)d_in[3];
    const float* net2_in = (const float*)d_in[4];
    const float* cz0 = (const float*)d_in[5];
    const float* cr0 = (const float*)d_in[6];
    const float* cq0 = (const float*)d_in[7];
    const float* cz1 = (const float*)d_in[8];
    const float* cr1 = (const float*)d_in[9];
    const float* cq1 = (const float*)d_in[10];
    const float* cz2 = (const float*)d_in[11];
    const float* cr2 = (const float*)d_in[12];
    const float* cq2 = (const float*)d_in[13];
#define P(i) ((const float*)d_in[(i)])
    const float *convc1_w = P(14), *convc1_b = P(15), *convc2_w = P(16), *convc2_b = P(17);
    const float *convf1_w = P(18), *convf1_b = P(19), *convf2_w = P(20), *convf2_b = P(21);
    const float *convm_w = P(22), *convm_b = P(23);
    const float *fh1_w = P(24), *fh1_b = P(25), *fh2_w = P(26), *fh2_b = P(27);
    const float *mk1_w = P(28), *mk1_b = P(29), *mk2_w = P(30), *mk2_b = P(31);
    const float *g08_wz = P(32), *g08_bz = P(33), *g08_wr = P(34), *g08_br = P(35), *g08_wq = P(36), *g08_bq = P(37);
    const float *g16_wz = P(38), *g16_bz = P(39), *g16_wr = P(40), *g16_br = P(41), *g16_wq = P(42), *g16_bq = P(43);
    const float *g32_wz = P(44), *g32_bz = P(45), *g32_wr = P(46), *g32_br = P(47), *g32_wq = P(48), *g32_bq = P(49);
#undef P

    // ---- workspace carve-out ----
    float* base = (float*)d_ws;
    size_t off = 0;
    auto alloc = [&](size_t n) { float* p = base + off; off += n; return p; };
    auto allocH = [&](size_t nh) {           // f16 alloc, 16B aligned
        off = (off + 3) & ~(size_t)3;
        _Float16* p = (_Float16*)(base + off);
        off += (nh + 1) / 2;
        return p;
    };
    const size_t NHW = (size_t)B * H * W;
    float* corr0 = alloc(NHW * W);
    float* corr1 = alloc(NHW * 36);
    float* corr2 = alloc(NHW * 18);
    float* corr3 = alloc(NHW * 9);
    float* cx    = alloc(NHW);
    float* netw0 = alloc((size_t)B * hd * H * W);
    float* netw1 = alloc((size_t)B * hd * H2 * W2);
    float* netw2 = alloc((size_t)B * hd * H4 * W4);
    float* corrf = alloc((size_t)B * 36 * H * W);
    float* flowb = alloc((size_t)B * 2 * H * W);
    float* cbuf1 = alloc((size_t)B * 64 * H * W);
    float* cbuf2 = alloc((size_t)B * 64 * H * W);
    float* fbuf1 = alloc((size_t)B * 64 * H * W);
    float* fbuf2 = alloc((size_t)B * 64 * H * W);
    float* mf    = alloc((size_t)B * 128 * H * W);
    float* pxA   = alloc((size_t)B * hd * H2 * W2);
    float* pxB   = alloc((size_t)B * hd * H4 * W4);
    float* it2   = alloc((size_t)B * hd * H2 * W2);
    float* it1   = alloc((size_t)B * hd * H * W);
    float* zb    = alloc((size_t)B * hd * H * W);
    float* rb    = alloc((size_t)B * hd * H * W);
    float* qb    = alloc((size_t)B * hd * H * W);
    float* rhb   = alloc((size_t)B * hd * H * W);
    float* fh1b  = alloc((size_t)B * 256 * H * W);
    float* dfb   = alloc((size_t)B * 2 * H * W);
    float* mk1b  = alloc((size_t)B * 256 * H * W);
    float* maskb = alloc((size_t)B * 576 * H * W);

    auto EW = [&](int n) { return dim3((n + 255) / 256); };
    // convert one weight tensor to padded f16
    auto cvtw = [&](const float* w, int Cout, int Ktot) {
        int Kpad = (Ktot + 31) & ~31;
        _Float16* p = allocH((size_t)Cout * Kpad);
        cvt_w_f16<<<EW(Cout * Kpad), 256, 0, stream>>>(w, p, Cout, Ktot, Kpad);
        return p;
    };
    const _Float16* convc1_h = cvtw(convc1_w, 64, 36);
    const _Float16* convc2_h = cvtw(convc2_w, 64, 576);
    const _Float16* convf1_h = cvtw(convf1_w, 64, 98);
    const _Float16* convf2_h = cvtw(convf2_w, 64, 576);
    const _Float16* convm_h  = cvtw(convm_w, 126, 1152);
    const _Float16* fh1_h    = cvtw(fh1_w, 256, 1152);
    const _Float16* fh2_h    = cvtw(fh2_w, 2, 2304);
    const _Float16* mk1_h    = cvtw(mk1_w, 256, 1152);
    const _Float16* mk2_h    = cvtw(mk2_w, 576, 256);
    const _Float16* g08_wz_h = cvtw(g08_wz, hd, 3456);
    const _Float16* g08_wr_h = cvtw(g08_wr, hd, 3456);
    const _Float16* g08_wq_h = cvtw(g08_wq, hd, 3456);
    const _Float16* g16_wz_h = cvtw(g16_wz, hd, 3456);
    const _Float16* g16_wr_h = cvtw(g16_wr, hd, 3456);
    const _Float16* g16_wq_h = cvtw(g16_wq, hd, 3456);
    const _Float16* g32_wz_h = cvtw(g32_wz, hd, 2304);
    const _Float16* g32_wr_h = cvtw(g32_wr, hd, 2304);
    const _Float16* g32_wq_h = cvtw(g32_wq, hd, 2304);

    auto conv = [&](const float* i0, int C0, const float* i1, int C1,
                    const float* i2, int C2, const _Float16* w, const float* bb,
                    const float* addv, float* o, int Hh, int Ww, int Cout,
                    int ostr, int ks, int pd, int act, float sc) {
        dim3 g((Cout + 63) / 64, (B * Hh * Ww + 63) / 64);
        conv_wmma<<<g, dim3(256), 0, stream>>>(i0, C0, i1, C1, i2, C2, w, bb, addv, o,
                                               B, Hh, Ww, Cout, ostr, ks, pd, act, sc);
    };

    // ---- one-time setup ----
    ew_tanh<<<EW(B * hd * H * W), 256, 0, stream>>>(net0_in, netw0, B * hd * H * W);
    ew_tanh<<<EW(B * hd * H2 * W2), 256, 0, stream>>>(net1_in, netw1, B * hd * H2 * W2);
    ew_tanh<<<EW(B * hd * H4 * W4), 256, 0, stream>>>(net2_in, netw2, B * hd * H4 * W4);
    {
        dim3 g(80 / 16, B * H);
        corr_wmma<80><<<g, dim3(256), 0, stream>>>(fmap1, fmap2, corr0, D, H, W);
    }
    avgpair<<<EW((int)(NHW * 36)), 256, 0, stream>>>(corr0, corr1, (int)NHW, 36);
    avgpair<<<EW((int)(NHW * 18)), 256, 0, stream>>>(corr1, corr2, (int)NHW, 18);
    avgpair<<<EW((int)(NHW * 9)), 256, 0, stream>>>(corr2, corr3, (int)NHW, 9);
    init_cx<<<EW((int)NHW), 256, 0, stream>>>(cx, (int)NHW, W);

    const int n08 = B * hd * H * W, n16 = B * hd * H2 * W2, n32 = B * hd * H4 * W4;

    for (int it = 0; it < ITERS; ++it) {
        corr_lookup_k<<<EW((int)NHW), 256, 0, stream>>>(corr0, corr1, corr2, corr3,
                                                        cx, corrf, B, H, W);
        flow_k<<<EW((int)NHW), 256, 0, stream>>>(cx, flowb, B, H, W);

        // ---- g32 ----
        pool2x_k<<<EW(n32), 256, 0, stream>>>(netw1, pxB, B, hd, H2, W2);
        conv(netw2, hd, pxB, hd, nullptr, 0, g32_wz_h, g32_bz, cz2, zb, H4, W4, hd, hd, 3, 1, 2, 1.f);
        conv(netw2, hd, pxB, hd, nullptr, 0, g32_wr_h, g32_br, cr2, rb, H4, W4, hd, hd, 3, 1, 2, 1.f);
        ew_mul<<<EW(n32), 256, 0, stream>>>(rb, netw2, rhb, n32);
        conv(rhb, hd, pxB, hd, nullptr, 0, g32_wq_h, g32_bq, cq2, qb, H4, W4, hd, hd, 3, 1, 3, 1.f);
        gru_comb<<<EW(n32), 256, 0, stream>>>(netw2, zb, qb, n32);

        // ---- g16 ----
        pool2x_k<<<EW(n16), 256, 0, stream>>>(netw0, pxA, B, hd, H, W);
        interp_k<<<EW(n16), 256, 0, stream>>>(netw2, it2, B, hd, H4, W4, H2, W2);
        conv(netw1, hd, pxA, hd, it2, hd, g16_wz_h, g16_bz, cz1, zb, H2, W2, hd, hd, 3, 1, 2, 1.f);
        conv(netw1, hd, pxA, hd, it2, hd, g16_wr_h, g16_br, cr1, rb, H2, W2, hd, hd, 3, 1, 2, 1.f);
        ew_mul<<<EW(n16), 256, 0, stream>>>(rb, netw1, rhb, n16);
        conv(rhb, hd, pxA, hd, it2, hd, g16_wq_h, g16_bq, cq1, qb, H2, W2, hd, hd, 3, 1, 3, 1.f);
        gru_comb<<<EW(n16), 256, 0, stream>>>(netw1, zb, qb, n16);

        // ---- motion encoder ----
        conv(corrf, 36, nullptr, 0, nullptr, 0, convc1_h, convc1_b, nullptr, cbuf1, H, W, 64, 64, 1, 0, 1, 1.f);
        conv(cbuf1, 64, nullptr, 0, nullptr, 0, convc2_h, convc2_b, nullptr, cbuf2, H, W, 64, 64, 3, 1, 1, 1.f);
        conv(flowb, 2, nullptr, 0, nullptr, 0, convf1_h, convf1_b, nullptr, fbuf1, H, W, 64, 64, 7, 3, 1, 1.f);
        conv(fbuf1, 64, nullptr, 0, nullptr, 0, convf2_h, convf2_b, nullptr, fbuf2, H, W, 64, 64, 3, 1, 1, 1.f);
        conv(cbuf2, 64, fbuf2, 64, nullptr, 0, convm_h, convm_b, nullptr, mf, H, W, 126, 128, 3, 1, 1, 1.f);
        mf_tail_k<<<EW((int)NHW), 256, 0, stream>>>(cx, mf, B, H, W);

        // ---- g08 ----
        interp_k<<<EW(n08), 256, 0, stream>>>(netw1, it1, B, hd, H2, W2, H, W);
        conv(netw0, hd, mf, 128, it1, hd, g08_wz_h, g08_bz, cz0, zb, H, W, hd, hd, 3, 1, 2, 1.f);
        conv(netw0, hd, mf, 128, it1, hd, g08_wr_h, g08_br, cr0, rb, H, W, hd, hd, 3, 1, 2, 1.f);
        ew_mul<<<EW(n08), 256, 0, stream>>>(rb, netw0, rhb, n08);
        conv(rhb, hd, mf, 128, it1, hd, g08_wq_h, g08_bq, cq0, qb, H, W, hd, hd, 3, 1, 3, 1.f);
        gru_comb<<<EW(n08), 256, 0, stream>>>(netw0, zb, qb, n08);

        // ---- flow head / mask head ----
        conv(netw0, hd, nullptr, 0, nullptr, 0, fh1_h, fh1_b, nullptr, fh1b, H, W, 256, 256, 3, 1, 1, 1.f);
        conv(fh1b, 256, nullptr, 0, nullptr, 0, fh2_h, fh2_b, nullptr, dfb, H, W, 2, 2, 3, 1, 0, 1.f);
        conv(netw0, hd, nullptr, 0, nullptr, 0, mk1_h, mk1_b, nullptr, mk1b, H, W, 256, 256, 3, 1, 1, 1.f);
        conv(mk1b, 256, nullptr, 0, nullptr, 0, mk2_h, mk2_b, nullptr, maskb, H, W, 576, 576, 1, 0, 0, 0.25f);

        coords_upd<<<EW((int)NHW), 256, 0, stream>>>(cx, dfb, B, H, W);

        float* outp = (float*)d_out + (size_t)it * B * (H * 8) * (W * 8);
        upsample_k<<<EW(B * H * 8 * W * 8), 256, 0, stream>>>(cx, maskb, outp, B, H, W);
    }
}
